// MultiStepGCNGRU_36069135352527
// MI455X (gfx1250) — matrix operand order repaired and checked by reading, not compiled, not run on hardware
//
#include <hip/hip_runtime.h>
#include <cstdint>

#define Bb   4
#define Pp   8
#define Nn   10000
#define Cc   64
#define Hh   256
#define Ee   160000
#define HOR  4
#define Mm   (Bb*Nn)   // 40000 rows

typedef __attribute__((ext_vector_type(4)))  __bf16 v4bf;
typedef __attribute__((ext_vector_type(8)))  __bf16 v8bf;
typedef __attribute__((ext_vector_type(16))) __bf16 v16bf;
typedef __attribute__((ext_vector_type(4)))  float  v4f;
typedef __attribute__((ext_vector_type(8)))  float  v8f;

// ----------------------------- utility kernels ------------------------------

__global__ void k_zero(float* p, int n) {
  int t = blockIdx.x * blockDim.x + threadIdx.x;
  if (t < n) p[t] = 0.f;
}

__global__ void k_count(const int* __restrict__ ei, float* cnt) {
  int t = blockIdx.x * blockDim.x + threadIdx.x;
  if (t < Ee) atomicAdd(&cnt[ei[Ee + t]], 1.f);   // row 1 = dst
}

__global__ void k_invert(float* cnt) {
  int t = blockIdx.x * blockDim.x + threadIdx.x;
  if (t < Nn) cnt[t] = 1.f / fmaxf(cnt[t], 1.f);
}

// Fused SAGE weight stored TRANSPOSED as Bt[n][k], n<256, k<192:
// k 0-63 = Wl0, 64-127 = Wl1, 128-191 = Wr0+Wr1.   Bias = bl0+bl1.
__global__ void k_pack_sage_wT(const float* __restrict__ Wl0, const float* __restrict__ Wl1,
                               const float* __restrict__ Wr0, const float* __restrict__ Wr1,
                               const float* __restrict__ bl0, const float* __restrict__ bl1,
                               __bf16* __restrict__ Wout, float* __restrict__ bout) {
  int t = blockIdx.x * blockDim.x + threadIdx.x;
  if (t < 256 * 192) {
    int n = t / 192, k = t % 192;
    float v;
    if      (k <  64) v = Wl0[k * 256 + n];
    else if (k < 128) v = Wl1[(k - 64) * 256 + n];
    else              v = Wr0[(k - 128) * 256 + n] + Wr1[(k - 128) * 256 + n];
    Wout[t] = (__bf16)v;
  }
  if (t < 256) bout[t] = bl0[t] + bl1[t];
}

// Plain fp32 -> bf16 copy. GRU Wih/Whh (768x256) and Wp (64x256) row-major are
// already the Bt[n][k] layout the GEMM wants (x @ W^T  ==  Bt = W).
__global__ void k_convert_bf16(const float* __restrict__ src, __bf16* __restrict__ dst, int n) {
  int t = blockIdx.x * blockDim.x + threadIdx.x;
  if (t < n) dst[t] = (__bf16)src[t];
}

// Scatter-add x[src] into agg[dst] over batched edges; one thread = 4 channels.
__global__ void k_scatter(const int* __restrict__ ei,
                          const float* __restrict__ xp, long xoff, long bstride,
                          float* __restrict__ agg) {
  long t = (long)blockIdx.x * blockDim.x + threadIdx.x;
  const long total = (long)Bb * Ee * 16;
  if (t >= total) return;
  int  q  = (int)(t & 15);
  long eg = t >> 4;
  int  b  = (int)(eg / Ee);
  int  e  = (int)(eg % Ee);
  int  src = ei[e];
  int  dst = ei[Ee + e];
  const float* xs = xp + xoff + (long)b * bstride + src * Cc + q * 4;
  float*       ad = agg + ((long)b * Nn + dst) * Cc + q * 4;
  atomicAdd(ad + 0, xs[0]);
  atomicAdd(ad + 1, xs[1]);
  atomicAdd(ad + 2, xs[2]);
  atomicAdd(ad + 3, xs[3]);
}

// Build SAGE GEMM input: Xc[m, 0:64]=mean0, [64:128]=mean1, [128:192]=x.
__global__ void k_pack_xc(const float* __restrict__ agg0, const float* __restrict__ agg1,
                          const float* __restrict__ inv0, const float* __restrict__ inv1,
                          const float* __restrict__ xp, long xoff, long bstride,
                          float* __restrict__ Xc) {
  long t = (long)blockIdx.x * blockDim.x + threadIdx.x;
  if (t >= (long)Mm * Cc) return;
  int  m = (int)(t >> 6);
  int  c = (int)(t & 63);
  int  b = m / Nn, n = m % Nn;
  float xv = xp[xoff + (long)b * bstride + n * Cc + c];
  Xc[m * 192 + c]       = agg0[m * Cc + c] * inv0[n];
  Xc[m * 192 + 64 + c]  = agg1[m * Cc + c] * inv1[n];
  Xc[m * 192 + 128 + c] = xv;
}

// GRU gate fusion (biases already added inside the two GEMMs).
__global__ void k_gate(const float* __restrict__ gi, const float* __restrict__ gh,
                       float* __restrict__ h) {
  int t = blockIdx.x * blockDim.x + threadIdx.x;
  if (t >= Mm * Hh) return;
  int m = t >> 8;
  int j = t & 255;
  const float* gim = gi + m * 768;
  const float* ghm = gh + m * 768;
  float r  = 1.f / (1.f + expf(-(gim[j]       + ghm[j])));
  float z  = 1.f / (1.f + expf(-(gim[256 + j] + ghm[256 + j])));
  float nn = tanhf(gim[512 + j] + r * ghm[512 + j]);
  float ho = h[t];
  h[t] = (1.f - z) * nn + z * ho;
}

// y[M,64] (row m = b*N+n) -> out[b][ts][n][c]
__global__ void k_writeout(const float* __restrict__ y, float* __restrict__ out, int ts) {
  int t = blockIdx.x * blockDim.x + threadIdx.x;
  if (t >= Mm * Cc) return;
  int m = t >> 6;
  int c = t & 63;
  int b = m / Nn, n = m % Nn;
  out[((b * HOR + ts) * Nn + n) * Cc + c] = y[t];
}

// ------------------------------- WMMA GEMM ----------------------------------
// D[M,N] = act(A[M,K](fp32) @ B[K,N] + bias[N]) with the weight given
// TRANSPOSED: Bt[n][k] bf16, so both LDS tiles are k-contiguous and every
// WMMA fragment is 1-2 ds_load_b128's.
// Block = 4 waves, 64x64 tile, DOUBLE-BUFFERED LDS: one barrier per K-slab,
// next slab's global loads issued before this slab's WMMAs.
__global__ __launch_bounds__(128)
void k_gemm(const float* __restrict__ A, const __bf16* __restrict__ Bt,
            const float* __restrict__ bias, float* __restrict__ D,
            int M, int N, int K, int relu) {
  __shared__ __bf16 As[2][64 * 32];   // [row][k], row pitch 64B
  __shared__ __bf16 Bs[2][64 * 32];   // [col][k], col pitch 64B

  const int tid  = threadIdx.x;
  const int wave = tid >> 5;
  const int lane = tid & 31;
  const int m0   = blockIdx.x * 64;
  const int n0   = blockIdx.y * 64;
  const int half = lane >> 4;      // 0: lanes 0-15, 1: lanes 16-31
  const int l16  = lane & 15;

  // staging coords: 2 threads per row, each covers 16 k's
  const int srow = tid >> 1;           // 0..63
  const int skc  = (tid & 1) * 16;     // 0 or 16

  const float*  Arow = A  + (m0 + srow) * K + skc;
  const __bf16* Brow = Bt + (n0 + srow) * K + skc;
  const int sIdx = srow * 32 + skc;

  v8f acc[4];
#pragma unroll
  for (int j = 0; j < 4; ++j)
#pragma unroll
    for (int i = 0; i < 8; ++i) acc[j][i] = 0.f;

  // One K-slab of WMMAs from LDS buffer `c`: batch all fragment loads, then
  // issue 4 independent back-to-back WMMAs.
  auto compute = [&](int c) {
    const __bf16* ap = &As[c][((wave << 4) + l16) * 32 + 16 * half];
    v8bf a0 = *(const v8bf*)ap;          // k = 16h .. 16h+7
    v8bf a1 = *(const v8bf*)(ap + 8);    // k = 16h+8 .. 16h+15
    v16bf a = __builtin_shufflevector(a0, a1, 0, 1, 2, 3, 4, 5, 6, 7,
                                      8, 9, 10, 11, 12, 13, 14, 15);
    v16bf b0 = *(const v16bf*)&Bs[c][(0 * 16 + l16) * 32 + 16 * half];
    v16bf b1 = *(const v16bf*)&Bs[c][(1 * 16 + l16) * 32 + 16 * half];
    v16bf b2 = *(const v16bf*)&Bs[c][(2 * 16 + l16) * 32 + 16 * half];
    v16bf b3 = *(const v16bf*)&Bs[c][(3 * 16 + l16) * 32 + 16 * half];
    acc[0] = __builtin_amdgcn_wmma_f32_16x16x32_bf16(false, a, false, b0, (short)0, acc[0], false, false);
    acc[1] = __builtin_amdgcn_wmma_f32_16x16x32_bf16(false, a, false, b1, (short)0, acc[1], false, false);
    acc[2] = __builtin_amdgcn_wmma_f32_16x16x32_bf16(false, a, false, b2, (short)0, acc[2], false, false);
    acc[3] = __builtin_amdgcn_wmma_f32_16x16x32_bf16(false, a, false, b3, (short)0, acc[3], false, false);
  };

  // ---- prologue: stage slab 0 into buffer 0 ----
#pragma unroll
  for (int u = 0; u < 4; ++u) {
    v4f f = *(const v4f*)(Arow + u * 4);
    v4bf h;
    h[0] = (__bf16)f[0]; h[1] = (__bf16)f[1];
    h[2] = (__bf16)f[2]; h[3] = (__bf16)f[3];
    *(v4bf*)&As[0][sIdx + u * 4] = h;
  }
#pragma unroll
  for (int u = 0; u < 2; ++u)
    *(v8bf*)&Bs[0][sIdx + u * 8] = *(const v8bf*)(Brow + u * 8);
  __syncthreads();

  // ---- pipelined main loop: one barrier per slab ----
  int cur = 0;
  for (int k0 = 32; k0 < K; k0 += 32) {
    // issue next slab's global loads first (overlap with WMMA below)
    v4f fa[4];
#pragma unroll
    for (int u = 0; u < 4; ++u) fa[u] = *(const v4f*)(Arow + k0 + u * 4);
    v8bf fb0 = *(const v8bf*)(Brow + k0);
    v8bf fb1 = *(const v8bf*)(Brow + k0 + 8);

    compute(cur);

    const int nxt = cur ^ 1;
#pragma unroll
    for (int u = 0; u < 4; ++u) {
      v4bf h;
      h[0] = (__bf16)fa[u][0]; h[1] = (__bf16)fa[u][1];
      h[2] = (__bf16)fa[u][2]; h[3] = (__bf16)fa[u][3];
      *(v4bf*)&As[nxt][sIdx + u * 4] = h;
    }
    *(v8bf*)&Bs[nxt][sIdx]     = fb0;
    *(v8bf*)&Bs[nxt][sIdx + 8] = fb1;
    __syncthreads();
    cur = nxt;
  }

  // ---- final slab ----
  compute(cur);

  // Epilogue: C/D layout — VGPR r -> row r + 8*half, lane%16 -> col.
#pragma unroll
  for (int j = 0; j < 4; ++j) {
    const int gcol = n0 + j * 16 + l16;
    const float bv = bias[gcol];
#pragma unroll
    for (int r = 0; r < 8; ++r) {
      const int grow = m0 + (wave << 4) + half * 8 + r;
      float v = acc[j][r] + bv;
      if (relu) v = fmaxf(v, 0.f);
      D[grow * N + gcol] = v;
    }
  }
}

// ------------------------------- launcher -----------------------------------

extern "C" void kernel_launch(void* const* d_in, const int* in_sizes, int n_in,
                              void* d_out, int out_size, void* d_ws, size_t ws_size,
                              hipStream_t stream) {
  (void)in_sizes; (void)n_in; (void)out_size; (void)ws_size;

  const float* x_seq = (const float*)d_in[0];
  const int*   ei0   = (const int*)d_in[1];
  const int*   ei1   = (const int*)d_in[2];
  const float* Wl0   = (const float*)d_in[3];
  const float* bl0   = (const float*)d_in[4];
  const float* Wr0   = (const float*)d_in[5];
  const float* Wl1   = (const float*)d_in[6];
  const float* bl1   = (const float*)d_in[7];
  const float* Wr1   = (const float*)d_in[8];
  const float* Wp    = (const float*)d_in[9];
  const float* bp    = (const float*)d_in[10];
  const float* gWih[3] = {(const float*)d_in[11], (const float*)d_in[15], (const float*)d_in[19]};
  const float* gWhh[3] = {(const float*)d_in[12], (const float*)d_in[16], (const float*)d_in[20]};
  const float* gbih[3] = {(const float*)d_in[13], (const float*)d_in[17], (const float*)d_in[21]};
  const float* gbhh[3] = {(const float*)d_in[14], (const float*)d_in[18], (const float*)d_in[22]};
  float* out = (float*)d_out;

  // ---- workspace carve-out (256B aligned) ----
  char*  w   = (char*)d_ws;
  size_t off = 0;
  auto take = [&](size_t bytes) -> char* {
    char* p = w + off;
    off = (off + bytes + 255) & ~(size_t)255;
    return p;
  };
  __bf16* wsage = (__bf16*)take((size_t)256 * 192 * 2);   // Bt layout [256][192]
  float*  bsage = (float*) take(256 * 4);
  __bf16* wih[3]; __bf16* whh[3];
  for (int g = 0; g < 3; ++g) {
    wih[g] = (__bf16*)take((size_t)768 * 256 * 2);        // Bt layout [768][256]
    whh[g] = (__bf16*)take((size_t)768 * 256 * 2);
  }
  __bf16* wpb = (__bf16*)take((size_t)64 * 256 * 2);      // Bt layout [64][256]
  float* inv0 = (float*)take((size_t)Nn * 4);
  float* inv1 = (float*)take((size_t)Nn * 4);
  float* agg0 = (float*)take((size_t)Mm * 64 * 4);
  float* agg1 = (float*)take((size_t)Mm * 64 * 4);
  float* Xc   = (float*)take((size_t)Mm * 192 * 4);
  float* cbuf = (float*)take((size_t)Mm * 256 * 4);
  float* gi   = (float*)take((size_t)Mm * 768 * 4);
  float* gh   = (float*)take((size_t)Mm * 768 * 4);
  float* hb[3];
  for (int g = 0; g < 3; ++g) hb[g] = (float*)take((size_t)Mm * 256 * 4);
  float* ybuf = (float*)take((size_t)Mm * 64 * 4);

  auto g1 = [](long n) { return dim3((unsigned)((n + 255) / 256)); };

  // ---- one-time weight prep + degree tables ----
  k_pack_sage_wT<<<g1(256 * 192), 256, 0, stream>>>(Wl0, Wl1, Wr0, Wr1, bl0, bl1, wsage, bsage);
  for (int g = 0; g < 3; ++g) {
    k_convert_bf16<<<g1(768 * 256), 256, 0, stream>>>(gWih[g], wih[g], 768 * 256);
    k_convert_bf16<<<g1(768 * 256), 256, 0, stream>>>(gWhh[g], whh[g], 768 * 256);
  }
  k_convert_bf16<<<g1(64 * 256), 256, 0, stream>>>(Wp, wpb, 64 * 256);

  k_zero<<<g1(Nn), 256, 0, stream>>>(inv0, Nn);
  k_zero<<<g1(Nn), 256, 0, stream>>>(inv1, Nn);
  k_count<<<g1(Ee), 256, 0, stream>>>(ei0, inv0);
  k_count<<<g1(Ee), 256, 0, stream>>>(ei1, inv1);
  k_invert<<<g1(Nn), 256, 0, stream>>>(inv0);
  k_invert<<<g1(Nn), 256, 0, stream>>>(inv1);

  for (int g = 0; g < 3; ++g)
    k_zero<<<g1(Mm * 256), 256, 0, stream>>>(hb[g], Mm * 256);

  // ---- time loop: P encoder steps + HORIZON decoder steps ----
  for (int t = 0; t < Pp + HOR; ++t) {
    const float* xp; long xoff, bstride;
    if (t < Pp)       { xp = x_seq; xoff = (long)t * Nn * Cc;        bstride = (long)Pp * Nn * Cc; }
    else if (t == Pp) { xp = x_seq; xoff = (long)(Pp - 1) * Nn * Cc; bstride = (long)Pp * Nn * Cc; }
    else              { xp = ybuf;  xoff = 0;                        bstride = (long)Nn * Cc; }

    k_zero<<<g1(Mm * 64), 256, 0, stream>>>(agg0, Mm * 64);
    k_zero<<<g1(Mm * 64), 256, 0, stream>>>(agg1, Mm * 64);
    k_scatter<<<g1((long)Bb * Ee * 16), 256, 0, stream>>>(ei0, xp, xoff, bstride, agg0);
    k_scatter<<<g1((long)Bb * Ee * 16), 256, 0, stream>>>(ei1, xp, xoff, bstride, agg1);
    k_pack_xc<<<g1(Mm * 64), 256, 0, stream>>>(agg0, agg1, inv0, inv1, xp, xoff, bstride, Xc);

    // Fused HeteroConv: c = relu(Xc[M,192] @ Wsage + bsage)
    {
      dim3 gg(Mm / 64, 256 / 64);
      k_gemm<<<gg, 128, 0, stream>>>(Xc, wsage, bsage, cbuf, Mm, 256, 192, 1);
    }

    // Three stacked GRUs
    const float* gin = cbuf;
    for (int g = 0; g < 3; ++g) {
      dim3 gg(Mm / 64, 768 / 64);
      k_gemm<<<gg, 128, 0, stream>>>(gin,   wih[g], gbih[g], gi, Mm, 768, 256, 0);
      k_gemm<<<gg, 128, 0, stream>>>(hb[g], whh[g], gbhh[g], gh, Mm, 768, 256, 0);
      k_gate<<<g1(Mm * 256), 256, 0, stream>>>(gi, gh, hb[g]);
      gin = hb[g];
    }

    // Decoder projection + output store (y also feeds the next step's input)
    if (t >= Pp) {
      dim3 gg(Mm / 64, 64 / 64);
      k_gemm<<<gg, 128, 0, stream>>>(hb[2], wpb, bp, ybuf, Mm, 64, 256, 0);
      k_writeout<<<g1(Mm * 64), 256, 0, stream>>>(ybuf, out, t - Pp);
    }
  }
}
